// PretrainingModel_63127429317073
// MI455X (gfx1250) — matrix operand order repaired and checked by reading, not compiled, not run on hardware
//
#include <hip/hip_runtime.h>

#define DEV __device__ __forceinline__

typedef __attribute__((ext_vector_type(16))) __bf16 v16bf;
typedef __attribute__((ext_vector_type(8)))  __bf16 v8bf;
typedef __attribute__((ext_vector_type(8)))  float  v8f;

constexpr int LAY = 12, DIM = 768, NH = 12, DHD = 64, VOC = 50265, FFD = 3072, BB = 2, SS = 1024;
constexpr int MM = BB * SS;       // 2048 token rows
constexpr int VPAD = 50304;       // VOC padded to multiple of 128 for async B tiles

// ---------- helpers ----------
DEV __bf16 f2bf(float f) {
  unsigned u = __builtin_bit_cast(unsigned, f);
  unsigned r = u + 0x7FFFu + ((u >> 16) & 1u);   // round-to-nearest-even
  unsigned short h = (unsigned short)(r >> 16);
  return __builtin_bit_cast(__bf16, h);
}

DEV unsigned lds_off(const void* p) { return (unsigned)(uintptr_t)p; }

// async 16B copy: global -> LDS, tracked by ASYNCcnt (CDNA5 GLOBAL_LOAD_ASYNC_TO_LDS_B128)
DEV void async_b128(unsigned ldsoff, const __bf16* g) {
  asm volatile("global_load_async_to_lds_b128 %0, %1, off"
               :: "v"(ldsoff), "v"(g) : "memory");
}

DEV void wait_async0() {
#if __has_builtin(__builtin_amdgcn_s_wait_asynccnt)
  __builtin_amdgcn_s_wait_asynccnt(0);
#else
  asm volatile("s_wait_asynccnt 0x0" ::: "memory");
#endif
}

DEV v16bf ld_frag(const __bf16* p0, const __bf16* p1) {
  v8bf a = *(const v8bf*)p0;
  v8bf b = *(const v8bf*)p1;
  return __builtin_shufflevector(a, b, 0,1,2,3,4,5,6,7,8,9,10,11,12,13,14,15);
}

DEV v8f wmma_bf16(v16bf a, v16bf b, v8f c) {
  return __builtin_amdgcn_wmma_f32_16x16x32_bf16(false, a, false, b, (short)0, c, false, false);
}

DEV float gelu_f(float x) { return 0.5f * x * (1.0f + erff(x * 0.70710678118654752f)); }

// ---------- weight convert: W[K][N] f32 -> Wt[N][K] bf16 (transpose via LDS) ----------
__global__ __launch_bounds__(256) void convT_kernel(const float* __restrict__ W,
                                                    __bf16* __restrict__ Wt, int K, int N) {
  __shared__ float tl[64][65];
  const int tid = threadIdx.x;
  const int n0 = blockIdx.x * 64, k0 = blockIdx.y * 64;
#pragma unroll
  for (int i = 0; i < 16; ++i) {
    const int e = tid + i * 256;
    const int r = e >> 6, c = e & 63;           // r: k-local, c: n-local
    tl[r][c] = W[(size_t)(k0 + r) * N + n0 + c];
  }
  __syncthreads();
#pragma unroll
  for (int i = 0; i < 16; ++i) {
    const int e = tid + i * 256;
    const int r = e >> 6, c = e & 63;           // r: n-local, c: k-local
    Wt[(size_t)(n0 + r) * K + k0 + c] = f2bf(tl[c][r]);
  }
}

// ---------- straight f32 -> bf16 with zero padding ----------
__global__ __launch_bounds__(256) void conv_pad_kernel(const float* __restrict__ in,
                                                       __bf16* __restrict__ out,
                                                       long n, long ntot) {
  long i = (long)blockIdx.x * 2048 + threadIdx.x;
#pragma unroll
  for (int j = 0; j < 8; ++j, i += 256)
    if (i < ntot) out[i] = (i < n) ? f2bf(in[i]) : f2bf(0.f);
}

// ---------- embedding gather ----------
__global__ __launch_bounds__(256) void embed_kernel(const float* __restrict__ we,
                                                    const float* __restrict__ pe,
                                                    const float* __restrict__ te,
                                                    const int* __restrict__ ids,
                                                    float* __restrict__ x) {
  const int row = blockIdx.x, tid = threadIdx.x;
  const int s = row & (SS - 1);
  const int id = ids[row];
#pragma unroll
  for (int i = 0; i < 3; ++i) {
    const int d = tid + i * 256;
    x[(size_t)row * DIM + d] = we[(size_t)id * DIM + d] + pe[(size_t)(s + 2) * DIM + d] + te[d];
  }
}

// ---------- (optional add) + LayerNorm; writes f32 AND bf16 outputs ----------
__global__ __launch_bounds__(256) void addln_kernel(const float* __restrict__ xin,
                                                    const float* __restrict__ tin, int hasT,
                                                    const float* __restrict__ g,
                                                    const float* __restrict__ bta,
                                                    float* __restrict__ outF,
                                                    __bf16* __restrict__ outB) {
  const int row = blockIdx.x, tid = threadIdx.x;
  float vv[3];
  float s = 0.f, ss = 0.f;
#pragma unroll
  for (int i = 0; i < 3; ++i) {
    const int d = tid + i * 256;
    float v = xin[(size_t)row * DIM + d];
    if (hasT) v += tin[(size_t)row * DIM + d];
    vv[i] = v; s += v; ss += v * v;
  }
  __shared__ float r1[256], r2[256];
  r1[tid] = s; r2[tid] = ss; __syncthreads();
  for (int o = 128; o > 0; o >>= 1) {
    if (tid < o) { r1[tid] += r1[tid + o]; r2[tid] += r2[tid + o]; }
    __syncthreads();
  }
  const float mean = r1[0] * (1.0f / DIM);
  const float var  = r2[0] * (1.0f / DIM) - mean * mean;
  const float inv  = rsqrtf(var + 1e-12f);
#pragma unroll
  for (int i = 0; i < 3; ++i) {
    const int d = tid + i * 256;
    const float o = (vv[i] - mean) * inv * g[d] + bta[d];
    outF[(size_t)row * DIM + d] = o;
    outB[(size_t)row * DIM + d] = f2bf(o);
  }
}

// ---------- WMMA GEMM, async double-buffered tiles ----------
// C[M,N] = A[M,K](bf16) x Wt[N,K](bf16, pre-transposed) (+bias, +gelu)
// EPI: 1=bias 2=bias+gelu.  OUTBF: 0 -> f32 Cf, 1 -> bf16 Cb.
template <int EPI, int OUTBF>
__global__ __launch_bounds__(256) void gemm_async_wmma(const __bf16* __restrict__ A,
                                                       const __bf16* __restrict__ Wt,
                                                       const float* __restrict__ bias,
                                                       float* __restrict__ Cf,
                                                       __bf16* __restrict__ Cb,
                                                       int Ndim, int Kdim) {
  __shared__ __attribute__((aligned(16))) __bf16 lA[2][64 * 40];
  __shared__ __attribute__((aligned(16))) __bf16 lB[2][128 * 40];
  const int tid = threadIdx.x;
  const int m0 = blockIdx.y * 64, n0 = blockIdx.x * 128;
  const int wave = tid >> 5, lane = tid & 31;
  const int wm = (wave >> 2) * 32, wn = (wave & 3) * 32;
  const int mrow = lane & 15, khalf = lane >> 4;

  // per-thread async copy assignments (16B chunks)
  const int arow = tid >> 2, ac = tid & 3;              // A: 64 rows x 4 chunks
  const int brow = tid >> 2, bc = tid & 3;              // B: 128 rows x 4 chunks (2 per thread)
  const __bf16* gA  = A  + (size_t)(m0 + arow) * Kdim + ac * 8;
  const __bf16* gB0 = Wt + (size_t)(n0 + brow) * Kdim + bc * 8;
  const __bf16* gB1 = Wt + (size_t)(n0 + 64 + brow) * Kdim + bc * 8;
  unsigned aoff[2], boff0[2], boff1[2];
#pragma unroll
  for (int b = 0; b < 2; ++b) {
    aoff[b]  = lds_off(&lA[b][arow * 40 + ac * 8]);
    boff0[b] = lds_off(&lB[b][brow * 40 + bc * 8]);
    boff1[b] = lds_off(&lB[b][(64 + brow) * 40 + bc * 8]);
  }

  const v8f vzero = {0.f, 0.f, 0.f, 0.f, 0.f, 0.f, 0.f, 0.f};
  v8f acc[2][2];
#pragma unroll
  for (int a = 0; a < 2; ++a)
#pragma unroll
    for (int b = 0; b < 2; ++b) acc[a][b] = vzero;

  const int nk = Kdim >> 5;
  // prologue: stage tile 0 into buffer 0
  async_b128(aoff[0], gA);
  async_b128(boff0[0], gB0);
  async_b128(boff1[0], gB1);

  int cur = 0;
  for (int kt = 0; kt < nk; ++kt) {
    wait_async0();
    __syncthreads();
    if (kt + 1 < nk) {               // prefetch next tile into other buffer
      const size_t ko = (size_t)(kt + 1) * 32;
      async_b128(aoff[cur ^ 1], gA + ko);
      async_b128(boff0[cur ^ 1], gB0 + ko);
      async_b128(boff1[cur ^ 1], gB1 + ko);
    }
    const __bf16* bA = &lA[cur][0];
    const __bf16* bB = &lB[cur][0];
    v16bf af[2], bf[2];
#pragma unroll
    for (int mi = 0; mi < 2; ++mi) {
      const __bf16* p = bA + (wm + mi * 16 + mrow) * 40 + khalf * 8;
      af[mi] = ld_frag(p, p + 16);
    }
#pragma unroll
    for (int ni = 0; ni < 2; ++ni) {
      const __bf16* p = bB + (wn + ni * 16 + mrow) * 40 + khalf * 8;
      bf[ni] = ld_frag(p, p + 16);
    }
#pragma unroll
    for (int mi = 0; mi < 2; ++mi)
#pragma unroll
      for (int ni = 0; ni < 2; ++ni) acc[mi][ni] = wmma_bf16(af[mi], bf[ni], acc[mi][ni]);
    cur ^= 1;
  }

#pragma unroll
  for (int mi = 0; mi < 2; ++mi)
#pragma unroll
    for (int ni = 0; ni < 2; ++ni) {
      const int gn = n0 + wn + ni * 16 + mrow;
      if (gn < Ndim) {
        const float bv = bias[gn];
#pragma unroll
        for (int r = 0; r < 8; ++r) {
          const int gm = m0 + wm + mi * 16 + khalf * 8 + r;
          float v = acc[mi][ni][r] + bv;
          if (EPI == 2) v = gelu_f(v);
          if (OUTBF) Cb[(size_t)gm * Ndim + gn] = f2bf(v);
          else       Cf[(size_t)gm * Ndim + gn] = v;
        }
      }
    }
}

// ---------- banded flash attention; bf16 in/out, async Q/K staging ----------
__global__ __launch_bounds__(128) void attn_kernel(const __bf16* __restrict__ qg,
                                                   const __bf16* __restrict__ kg,
                                                   const __bf16* __restrict__ vg,
                                                   const int* __restrict__ amask,
                                                   __bf16* __restrict__ ctx, int w2) {
  __shared__ __attribute__((aligned(16))) __bf16 lQ[64 * 72];
  __shared__ __attribute__((aligned(16))) __bf16 lK[64 * 72];
  __shared__ __attribute__((aligned(16))) __bf16 lVt[64 * 72];   // [dh][key]
  __shared__ __attribute__((aligned(16))) __bf16 lP[4][16 * 72];
  __shared__ float lBad[64];
  const int tid = threadIdx.x;
  const int qt = blockIdx.x, h = blockIdx.y, b = blockIdx.z;
  const int bS = b * SS;
  const int hoff = h * DHD;
  const int q0 = qt * 64;
  const int wave = tid >> 5, lane = tid & 31;
  const int mrow = lane & 15, khalf = lane >> 4;

  // stage Q asynchronously (64 rows x 8 chunks of 16B)
#pragma unroll
  for (int i = 0; i < 4; ++i) {
    const int e = tid + i * 128;
    const int r = e >> 3, cc = e & 7;
    async_b128(lds_off(&lQ[r * 72 + cc * 8]),
               qg + (size_t)(bS + q0 + r) * DIM + hoff + cc * 8);
  }

  const v8f vzero = {0.f, 0.f, 0.f, 0.f, 0.f, 0.f, 0.f, 0.f};
  v8f acc[4];
  float mi_r[8], li_r[8];
#pragma unroll
  for (int j = 0; j < 4; ++j) acc[j] = vzero;
#pragma unroll
  for (int r = 0; r < 8; ++r) { mi_r[r] = -1e30f; li_r[r] = 0.f; }

  int klo = q0 - w2; if (klo < 0) klo = 0;
  int khi = q0 + 63 + w2; if (khi > SS - 1) khi = SS - 1;
  const int ktlo = klo >> 6, kthi = khi >> 6;

  for (int kt = ktlo; kt <= kthi; ++kt) {
    const int k0 = kt * 64;
    __syncthreads();   // previous tile fully consumed
    // K tile: async
#pragma unroll
    for (int i = 0; i < 4; ++i) {
      const int e = tid + i * 128;
      const int r = e >> 3, cc = e & 7;
      async_b128(lds_off(&lK[r * 72 + cc * 8]),
                 kg + (size_t)(bS + k0 + r) * DIM + hoff + cc * 8);
    }
    // V tile: load bf16 + transpose into LDS
#pragma unroll
    for (int i = 0; i < 4; ++i) {
      const int e = tid + i * 128;
      const int r = e >> 3, cc = e & 7;
      v8bf vv = *(const v8bf*)(vg + (size_t)(bS + k0 + r) * DIM + hoff + cc * 8);
#pragma unroll
      for (int j = 0; j < 8; ++j) lVt[(cc * 8 + j) * 72 + r] = vv[j];
    }
    if (tid < 64) lBad[tid] = (amask[bS + k0 + tid] > 0) ? 0.f : 1.f;
    wait_async0();
    __syncthreads();

    // S = Q K^T
    v8f s[4];
#pragma unroll
    for (int j = 0; j < 4; ++j) s[j] = vzero;
#pragma unroll
    for (int ks = 0; ks < 2; ++ks) {
      const __bf16* qa = &lQ[(wave * 16 + mrow) * 72 + ks * 32 + khalf * 8];
      v16bf af = ld_frag(qa, qa + 16);
#pragma unroll
      for (int j = 0; j < 4; ++j) {
        const __bf16* kp = &lK[(j * 16 + mrow) * 72 + ks * 32 + khalf * 8];
        v16bf bf = ld_frag(kp, kp + 16);
        s[j] = wmma_bf16(af, bf, s[j]);
      }
    }

    // scale + band/key mask (reference's -1e9 convention)
    float sv[4][8];
#pragma unroll
    for (int j = 0; j < 4; ++j) {
      const int kpos = k0 + j * 16 + mrow;
      const float bad = lBad[j * 16 + mrow];
#pragma unroll
      for (int r = 0; r < 8; ++r) {
        const int qpos = q0 + wave * 16 + khalf * 8 + r;
        const int dd = qpos >= kpos ? qpos - kpos : kpos - qpos;
        const bool ok = (dd <= w2) && (bad == 0.f);
        sv[j][r] = ok ? s[j][r] * 0.125f : -1e9f;
      }
    }

    // online softmax; each row spans 16 lanes (halves independent)
    float corr[8];
#pragma unroll
    for (int r = 0; r < 8; ++r) {
      float t = fmaxf(fmaxf(sv[0][r], sv[1][r]), fmaxf(sv[2][r], sv[3][r]));
      t = fmaxf(t, __shfl_xor(t, 1, 16));
      t = fmaxf(t, __shfl_xor(t, 2, 16));
      t = fmaxf(t, __shfl_xor(t, 4, 16));
      t = fmaxf(t, __shfl_xor(t, 8, 16));
      const float mnew = fmaxf(mi_r[r], t);
      corr[r] = __expf(mi_r[r] - mnew);
      float su = 0.f;
#pragma unroll
      for (int j = 0; j < 4; ++j) {
        const float p = __expf(sv[j][r] - mnew);
        sv[j][r] = p; su += p;
      }
      su += __shfl_xor(su, 1, 16);
      su += __shfl_xor(su, 2, 16);
      su += __shfl_xor(su, 4, 16);
      su += __shfl_xor(su, 8, 16);
      li_r[r] = li_r[r] * corr[r] + su;
      mi_r[r] = mnew;
    }
#pragma unroll
    for (int j = 0; j < 4; ++j)
#pragma unroll
      for (int r = 0; r < 8; ++r) acc[j][r] = acc[j][r] * corr[r];

    // stage P (C-layout regs -> A-layout LDS), per-wave region
    __bf16* pb = &lP[wave][0];
#pragma unroll
    for (int j = 0; j < 4; ++j)
#pragma unroll
      for (int r = 0; r < 8; ++r)
        pb[(khalf * 8 + r) * 72 + j * 16 + mrow] = f2bf(sv[j][r]);

    // O += P V
#pragma unroll
    for (int ks = 0; ks < 2; ++ks) {
      const __bf16* pa = &lP[wave][mrow * 72 + ks * 32 + khalf * 8];
      v16bf af = ld_frag(pa, pa + 16);
#pragma unroll
      for (int j = 0; j < 4; ++j) {
        const __bf16* vp = &lVt[(j * 16 + mrow) * 72 + ks * 32 + khalf * 8];
        v16bf bf = ld_frag(vp, vp + 16);
        acc[j] = wmma_bf16(af, bf, acc[j]);
      }
    }
  }

#pragma unroll
  for (int j = 0; j < 4; ++j)
#pragma unroll
    for (int r = 0; r < 8; ++r) {
      const float o = acc[j][r] / li_r[r];
      const size_t row = (size_t)(bS + q0 + wave * 16 + khalf * 8 + r);
      ctx[row * DIM + hoff + j * 16 + mrow] = f2bf(o);
    }
}

// ---------- MLM loss ----------
__global__ __launch_bounds__(256) void loss_kernel(const float* __restrict__ logits,
                                                   const int* __restrict__ labels,
                                                   float* __restrict__ nllb,
                                                   float* __restrict__ cntb) {
  const int row = blockIdx.x, tid = threadIdx.x;
  __shared__ float red[256];
  const float* lr = logits + (size_t)row * VOC;
  float mx = -3.0e38f;
  for (int i = tid; i < VOC; i += 256) mx = fmaxf(mx, lr[i]);
  red[tid] = mx; __syncthreads();
  for (int o = 128; o > 0; o >>= 1) { if (tid < o) red[tid] = fmaxf(red[tid], red[tid + o]); __syncthreads(); }
  mx = red[0]; __syncthreads();
  float se = 0.f;
  for (int i = tid; i < VOC; i += 256) se += __expf(lr[i] - mx);
  red[tid] = se; __syncthreads();
  for (int o = 128; o > 0; o >>= 1) { if (tid < o) red[tid] += red[tid + o]; __syncthreads(); }
  se = red[0];
  if (tid == 0) {
    const int lab = labels[row];
    if (lab >= 0) { nllb[row] = -(lr[lab] - mx - logf(se)); cntb[row] = 1.f; }
    else          { nllb[row] = 0.f; cntb[row] = 0.f; }
  }
}

__global__ __launch_bounds__(256) void reduce_loss_kernel(const float* __restrict__ nllb,
                                                          const float* __restrict__ cntb,
                                                          float* __restrict__ out) {
  __shared__ float r1[256], r2[256];
  const int tid = threadIdx.x;
  float a = 0.f, c = 0.f;
  for (int i = tid; i < MM; i += 256) { a += nllb[i]; c += cntb[i]; }
  r1[tid] = a; r2[tid] = c; __syncthreads();
  for (int o = 128; o > 0; o >>= 1) { if (tid < o) { r1[tid] += r1[tid + o]; r2[tid] += r2[tid + o]; } __syncthreads(); }
  if (tid == 0) out[0] = r1[0] / fmaxf(r2[0], 1.f);
}

// ---------- launch ----------
extern "C" void kernel_launch(void* const* d_in, const int* in_sizes, int n_in,
                              void* d_out, int out_size, void* d_ws, size_t ws_size,
                              hipStream_t stream) {
  (void)in_sizes; (void)n_in; (void)out_size; (void)ws_size;
  const float* word_emb = (const float*)d_in[0];
  const float* pos_emb  = (const float*)d_in[1];
  const float* tok_emb  = (const float*)d_in[2];
  const float* eg = (const float*)d_in[3];
  const float* eb = (const float*)d_in[4];
  const float* Wq = (const float*)d_in[5];
  const float* bq = (const float*)d_in[6];
  const float* Wk = (const float*)d_in[7];
  const float* bk = (const float*)d_in[8];
  const float* Wv = (const float*)d_in[9];
  const float* bv = (const float*)d_in[10];
  const float* Wo = (const float*)d_in[11];
  const float* bo = (const float*)d_in[12];
  const float* ag = (const float*)d_in[13];
  const float* ab = (const float*)d_in[14];
  const float* W1 = (const float*)d_in[15];
  const float* b1 = (const float*)d_in[16];
  const float* W2 = (const float*)d_in[17];
  const float* b2 = (const float*)d_in[18];
  const float* fg = (const float*)d_in[19];
  const float* fb = (const float*)d_in[20];
  const float* ldw = (const float*)d_in[21];
  const float* ldb = (const float*)d_in[22];
  const float* lg  = (const float*)d_in[23];
  const float* lb  = (const float*)d_in[24];
  const float* lm_bias = (const float*)d_in[25];
  const int* ids    = (const int*)d_in[26];
  const int* amask  = (const int*)d_in[27];
  const int* labels = (const int*)d_in[28];

  const size_t MD = (size_t)MM * DIM;
  const size_t DD = (size_t)DIM * DIM;
  const size_t DF = (size_t)DIM * FFD;

  // f32 region
  float* x    = (float*)d_ws;
  float* t1   = x + MD;
  float* nllb = t1 + MD;
  float* cntb = nllb + MM;
  // bf16 region (starts 16B-aligned)
  __bf16* xbf   = (__bf16*)(cntb + MM);
  __bf16* qbf   = xbf + MD;
  __bf16* kbf   = qbf + MD;
  __bf16* vbf   = kbf + MD;
  __bf16* ctxbf = vbf + MD;
  __bf16* t1bf  = ctxbf + MD;
  __bf16* hffbf = t1bf + MD;
  __bf16* wqT   = hffbf + (size_t)MM * FFD;   // per-layer staging
  __bf16* wkT   = wqT + DD;
  __bf16* wvT   = wkT + DD;
  __bf16* woT   = wvT + DD;
  __bf16* w1T   = woT + DD;                   // [FFD][DIM]
  __bf16* w2T   = w1T + DF;                   // [DIM][FFD]
  __bf16* ldwT  = w2T + DF;
  __bf16* wembf = ldwT + DD;                  // [VPAD][DIM]

  float* outF = (float*)d_out;
  float* logits = outF + 1;

  const dim3 blk(256);
  const dim3 gD((DIM + 127) / 128, MM / 64);
  const dim3 gF((FFD + 127) / 128, MM / 64);
  const dim3 gV((VPAD + 127) / 128, MM / 64);
  const dim3 gTdd(DIM / 64, DIM / 64);
  const dim3 gTdf(FFD / 64, DIM / 64);   // W1: K=DIM rows, N=FFD cols
  const dim3 gTfd(DIM / 64, FFD / 64);   // W2: K=FFD rows, N=DIM cols
  static const int WIN[LAY] = {32, 32, 64, 64, 128, 128, 256, 256, 512, 512, 512, 512};

  // one-time conversions
  {
    const long nwe = (long)VOC * DIM, nwep = (long)VPAD * DIM;
    conv_pad_kernel<<<(unsigned)((nwep + 2047) / 2048), blk, 0, stream>>>(word_emb, wembf, nwe, nwep);
    convT_kernel<<<gTdd, blk, 0, stream>>>(ldw, ldwT, DIM, DIM);
  }

  embed_kernel<<<MM, blk, 0, stream>>>(word_emb, pos_emb, tok_emb, ids, x);
  addln_kernel<<<MM, blk, 0, stream>>>(x, x, 0, eg, eb, x, xbf);

  for (int l = 0; l < LAY; ++l) {
    const size_t oDD = (size_t)l * DD;
    convT_kernel<<<gTdd, blk, 0, stream>>>(Wq + oDD, wqT, DIM, DIM);
    convT_kernel<<<gTdd, blk, 0, stream>>>(Wk + oDD, wkT, DIM, DIM);
    convT_kernel<<<gTdd, blk, 0, stream>>>(Wv + oDD, wvT, DIM, DIM);
    convT_kernel<<<gTdd, blk, 0, stream>>>(Wo + oDD, woT, DIM, DIM);
    convT_kernel<<<gTdf, blk, 0, stream>>>(W1 + (size_t)l * DF, w1T, DIM, FFD);
    convT_kernel<<<gTfd, blk, 0, stream>>>(W2 + (size_t)l * DF, w2T, FFD, DIM);

    gemm_async_wmma<1, 1><<<gD, blk, 0, stream>>>(xbf, wqT, bq + l * DIM, nullptr, qbf, DIM, DIM);
    gemm_async_wmma<1, 1><<<gD, blk, 0, stream>>>(xbf, wkT, bk + l * DIM, nullptr, kbf, DIM, DIM);
    gemm_async_wmma<1, 1><<<gD, blk, 0, stream>>>(xbf, wvT, bv + l * DIM, nullptr, vbf, DIM, DIM);
    attn_kernel<<<dim3(SS / 64, NH, BB), dim3(128), 0, stream>>>(qbf, kbf, vbf, amask, ctxbf, WIN[l] / 2);
    gemm_async_wmma<1, 0><<<gD, blk, 0, stream>>>(ctxbf, woT, bo + l * DIM, t1, nullptr, DIM, DIM);
    addln_kernel<<<MM, blk, 0, stream>>>(x, t1, 1, ag + l * DIM, ab + l * DIM, x, xbf);
    gemm_async_wmma<2, 1><<<gF, blk, 0, stream>>>(xbf, w1T, b1 + l * FFD, nullptr, hffbf, FFD, DIM);
    gemm_async_wmma<1, 0><<<gD, blk, 0, stream>>>(hffbf, w2T, b2 + l * DIM, t1, nullptr, DIM, FFD);
    addln_kernel<<<MM, blk, 0, stream>>>(x, t1, 1, fg + l * DIM, fb + l * DIM, x, xbf);
  }

  // LM head
  gemm_async_wmma<2, 0><<<gD, blk, 0, stream>>>(xbf, ldwT, ldb, t1, nullptr, DIM, DIM);
  addln_kernel<<<MM, blk, 0, stream>>>(t1, t1, 0, lg, lb, t1, t1bf);
  gemm_async_wmma<1, 0><<<gV, blk, 0, stream>>>(t1bf, wembf, lm_bias, logits, nullptr, VOC, DIM);

  loss_kernel<<<MM, blk, 0, stream>>>(logits, labels, nllb, cntb);
  reduce_loss_kernel<<<1, blk, 0, stream>>>(nllb, cntb, outF);
}
